// NodeSchNetWrapper_44564580663334
// MI455X (gfx1250) — matrix operand order repaired and checked by reading, not compiled, not run on hardware
//
#include <hip/hip_runtime.h>
#include <hip/hip_bf16.h>

// ---------------- problem constants (from the reference) ----------------
#define HID   600          // hidden channels
#define HP    640          // padded hidden (mult of 64 for 4-subtile waves)
#define KP_H  640          // padded K extent for K=600 (mult of 64: even #ksteps)
#define NGAU  50           // gaussians
#define KP_G  64           // padded gaussian K (2 ksteps, even)
#define NIter 6
#define NPER  24           // nodes per graph
#define GG    128          // graphs
#define NN    (NPER*GG)    // 3072 nodes
#define EPG   (NPER*(NPER-1)) // 552 edges per graph
#define EE    (GG*EPG)     // 70656 edges
#define NSUB  4            // n-tiles per wave (16x64 output tile per wave)

typedef _Float16 h8v  __attribute__((ext_vector_type(8)));
typedef _Float16 v16h __attribute__((ext_vector_type(16)));
typedef float    v8f  __attribute__((ext_vector_type(8)));

__device__ __forceinline__ float ssp(float x) {
  const float LN2 = 0.6931471805599453f;
  return (x > 20.f) ? (x - LN2) : (log1pf(__expf(x)) - LN2);
}

// A fragment (16x32 f16): lanes0-15 hold K 0-7 / 16-23 ; lanes16-31 K 8-15 / 24-31
__device__ __forceinline__ v16h load_a_frag(const _Float16* __restrict__ Arow,
                                            int k0, int hs) {
  h8v lo = *(const h8v*)(Arow + k0 + hs * 8);
  h8v hi = *(const h8v*)(Arow + k0 + 16 + hs * 8);
  v16h a;
  #pragma unroll
  for (int i = 0; i < 8; ++i) { a[i] = lo[i]; a[8 + i] = hi[i]; }
  return a;
}

// B fragment (32x16 f16): lanes0-15 K 0-7 / 16-23 ; lanes16-31 K 8-15 / 24-31 (col = lane%16)
__device__ __forceinline__ v16h load_b_frag(const _Float16* __restrict__ Bcol,
                                            int k0, int hs) {
  h8v lo = *(const h8v*)(Bcol + k0 + hs * 8);
  h8v hi = *(const h8v*)(Bcol + k0 + 16 + hs * 8);
  v16h b;
  #pragma unroll
  for (int i = 0; i < 8; ++i) { b[i] = lo[i]; b[8 + i] = hi[i]; }
  return b;
}

// ---------------- weight convert: fp32 [K,N] -> f16 transposed padded [NP][KP]
__global__ __launch_bounds__(256)
void convert_wT_kernel(const float* __restrict__ W, _Float16* __restrict__ Bt,
                       int K, int N, int KP, int NP) {
  int idx = blockIdx.x * blockDim.x + threadIdx.x;
  if (idx >= NP * KP) return;
  int n = idx / KP, k = idx - n * KP;
  float v = (k < K && n < N) ? W[(size_t)k * N + n] : 0.f;
  Bt[idx] = (_Float16)v;
}

// ---------------- edge features: dist -> gaussian smearing (f16) + cosine cutoff
__global__ __launch_bounds__(256)
void edge_feat_kernel(const float* __restrict__ pos, const int* __restrict__ ei,
                      _Float16* __restrict__ eF, float* __restrict__ Cc) {
  int e = blockIdx.x * blockDim.x + threadIdx.x;
  if (e >= EE) return;
  int s = ei[e], d = ei[EE + e];
  float dx = pos[3 * s + 0] - pos[3 * d + 0];
  float dy = pos[3 * s + 1] - pos[3 * d + 1];
  float dz = pos[3 * s + 2] - pos[3 * d + 2];
  float dist = sqrtf(dx * dx + dy * dy + dz * dz);
  Cc[e] = 0.5f * (cosf(dist * 0.31415926535f) + 1.f);   // pi/10
  const float step  = 10.f / 49.f;
  const float coeff = -0.5f / (step * step);
  _Float16* row = eF + (size_t)e * KP_G;
  #pragma unroll 1
  for (int g = 0; g < NGAU; ++g) {
    float dd = dist - (float)g * step;
    row[g] = (_Float16)__expf(coeff * dd * dd);
  }
  for (int g = NGAU; g < KP_G; ++g) row[g] = (_Float16)0.f;
}

// ---------------- embedding gather -> h32 [N,600] and hF [N,HP] (zero-padded)
__global__ __launch_bounds__(256)
void embed_kernel(const int* __restrict__ z, const float* __restrict__ emb,
                  float* __restrict__ h32, _Float16* __restrict__ hF) {
  int idx = blockIdx.x * blockDim.x + threadIdx.x;
  if (idx >= NN * HP) return;
  int n = idx / HP, c = idx - n * HP;
  float v = 0.f;
  if (c < HID) {
    v = emb[(size_t)z[n] * HID + c];
    h32[(size_t)n * HID + c] = v;
  }
  hF[idx] = (_Float16)v;
}

// ---------------- generic f16 WMMA GEMM: out = epilogue(A[M,K] * Bt^T + ...)
// A: f16 row-major, lda halfs (zero-padded to K).  Bt: f16 [col][k], ldb halfs.
// Each wave computes a 16x64 tile (4 WMMA accumulators). The K loop is manually
// unrolled 2x with ping-pong fragment buffers: loads for step k+32 issue before
// the WMMAs of step k, giving period-1 loop-carried values (no register copies)
// and partial s_wait_loadcnt overlap of HBM latency with the matrix pipe.
// Requires: NP mult of 64, M mult of 16, K mult of 64 (even #ksteps).
// Epilogue: +bias, ssp, *rowscale, +residual, dual f32(ld=N)/f16(ld=NP) stores.
__global__ __launch_bounds__(256)
void gemm_wmma_kernel(const _Float16* __restrict__ A, int lda,
                      const _Float16* __restrict__ Bt, int ldb,
                      const float* __restrict__ bias,
                      const float* __restrict__ rowscale,
                      const float* __restrict__ residual,
                      float* __restrict__ out32,
                      _Float16* __restrict__ out16,
                      int M, int N, int NP, int K, int act) {
  const int NG4 = NP >> 6;                       // groups of 4 n-tiles
  const int groups = (M >> 4) * NG4;
  int wid = blockIdx.x * (blockDim.x >> 5) + (threadIdx.x >> 5);
  if (wid >= groups) return;                     // whole-wave exit: EXEC stays all-1
  int mT = wid / NG4, nG = wid - mT * NG4;
  int lane = threadIdx.x & 31;
  int hs = lane >> 4, l = lane & 15;

  const _Float16* Arow = A + (size_t)(mT * 16 + l) * lda;
  const _Float16* Bp[NSUB];
  #pragma unroll
  for (int s = 0; s < NSUB; ++s)
    Bp[s] = Bt + (size_t)(nG * 64 + s * 16 + l) * ldb;

  v8f acc[NSUB];
  #pragma unroll
  for (int s = 0; s < NSUB; ++s)
    acc[s] = (v8f){0.f, 0.f, 0.f, 0.f, 0.f, 0.f, 0.f, 0.f};

  // ---- 2x-unrolled ping-pong K loop (even number of 32-wide ksteps) ----
  v16h a0 = load_a_frag(Arow, 0, hs);
  v16h b0[NSUB];
  #pragma unroll
  for (int s = 0; s < NSUB; ++s) b0[s] = load_b_frag(Bp[s], 0, hs);

  for (int k0 = 0; k0 + 64 < K; k0 += 64) {
    v16h a1 = load_a_frag(Arow, k0 + 32, hs);
    v16h b1[NSUB];
    #pragma unroll
    for (int s = 0; s < NSUB; ++s) b1[s] = load_b_frag(Bp[s], k0 + 32, hs);
    #pragma unroll
    for (int s = 0; s < NSUB; ++s)
      acc[s] = __builtin_amdgcn_wmma_f32_16x16x32_f16(
          false, a0, false, b0[s], (short)0, acc[s], false, false);
    a0 = load_a_frag(Arow, k0 + 64, hs);
    #pragma unroll
    for (int s = 0; s < NSUB; ++s) b0[s] = load_b_frag(Bp[s], k0 + 64, hs);
    #pragma unroll
    for (int s = 0; s < NSUB; ++s)
      acc[s] = __builtin_amdgcn_wmma_f32_16x16x32_f16(
          false, a1, false, b1[s], (short)0, acc[s], false, false);
  }
  {  // peeled final pair of ksteps: K-64 (in a0/b0) and K-32
    v16h a1 = load_a_frag(Arow, K - 32, hs);
    v16h b1[NSUB];
    #pragma unroll
    for (int s = 0; s < NSUB; ++s) b1[s] = load_b_frag(Bp[s], K - 32, hs);
    #pragma unroll
    for (int s = 0; s < NSUB; ++s)
      acc[s] = __builtin_amdgcn_wmma_f32_16x16x32_f16(
          false, a0, false, b0[s], (short)0, acc[s], false, false);
    #pragma unroll
    for (int s = 0; s < NSUB; ++s)
      acc[s] = __builtin_amdgcn_wmma_f32_16x16x32_f16(
          false, a1, false, b1[s], (short)0, acc[s], false, false);
  }

  // ---- epilogue ----
  int mBase = mT * 16;
  #pragma unroll
  for (int s = 0; s < NSUB; ++s) {
    int col = nG * 64 + s * 16 + l;
    bool cin = (col < N);
    float bv = (bias && cin) ? bias[col] : 0.f;
    #pragma unroll
    for (int r = 0; r < 8; ++r) {
      int orow = mBase + hs * 8 + r;              // C layout: vgpr r -> row, lane -> col
      float v = acc[s][r] + bv;
      if (act) v = ssp(v);
      if (rowscale) v *= rowscale[orow];
      if (residual && cin) v += residual[(size_t)orow * N + col];
      if (out32 && cin) out32[(size_t)orow * N + col] = v;
      if (out16) out16[(size_t)orow * (size_t)NP + col] = cin ? (_Float16)v : (_Float16)0.f;
    }
  }
}

// ---------------- scatter-free aggregation using the known complete-graph layout
// agg[n,c] = sum_{i != j} x[g*24+i, c] * Wt[edge(g,i,j), c]
__global__ __launch_bounds__(256)
void aggregate_kernel(const float* __restrict__ x32,
                      const _Float16* __restrict__ WtF,
                      _Float16* __restrict__ aggF) {
  int idx = blockIdx.x * blockDim.x + threadIdx.x;
  if (idx >= NN * HP) return;
  int n = idx / HP, c = idx - n * HP;
  if (c >= HID) { aggF[idx] = (_Float16)0.f; return; }
  int g = n / NPER, j = n - g * NPER;
  const float*    xg = x32 + (size_t)g * NPER * HID + c;
  const _Float16* wg = WtF + (size_t)g * EPG * HP + c;
  float acc = 0.f;
  #pragma unroll
  for (int i = 0; i < NPER; ++i) {
    if (i == j) continue;
    int le = i * (NPER - 1) + (j < i ? j : j - 1);   // edge order from meshgrid(i,j)
    acc += xg[(size_t)i * HID] * (float)wg[(size_t)le * HP];
  }
  aggF[idx] = (_Float16)acc;
}

// ---------------- mean pool over the 24 nodes of each graph -> f16 padded
__global__ __launch_bounds__(256)
void pool_kernel(const float* __restrict__ h32, _Float16* __restrict__ pooledF) {
  int idx = blockIdx.x * blockDim.x + threadIdx.x;
  if (idx >= GG * HP) return;
  int g = idx / HP, c = idx - g * HP;
  if (c >= HID) { pooledF[idx] = (_Float16)0.f; return; }
  const float* base = h32 + (size_t)g * NPER * HID + c;
  float s = 0.f;
  #pragma unroll
  for (int j = 0; j < NPER; ++j) s += base[(size_t)j * HID];
  pooledF[idx] = (_Float16)(s * (1.f / (float)NPER));
}

// ======================================================================
extern "C" void kernel_launch(void* const* d_in, const int* in_sizes, int n_in,
                              void* d_out, int out_size, void* d_ws, size_t ws_size,
                              hipStream_t stream) {
  const int*   z      = (const int*)  d_in[0];
  const float* pos    = (const float*)d_in[1];
  // d_in[2] = batch (derived from node index instead)
  const int*   ei     = (const int*)  d_in[3];
  const float* emb    = (const float*)d_in[4];
  const float* mlp_w1 = (const float*)d_in[5];
  const float* mlp_b1 = (const float*)d_in[6];
  const float* mlp_w2 = (const float*)d_in[7];
  const float* mlp_b2 = (const float*)d_in[8];
  const float* lin1_w = (const float*)d_in[9];
  const float* lin2_w = (const float*)d_in[10];
  const float* lin2_b = (const float*)d_in[11];
  const float* lin_w  = (const float*)d_in[12];
  const float* lin_b  = (const float*)d_in[13];
  const float* pool_w = (const float*)d_in[14];
  const float* pool_b = (const float*)d_in[15];
  float* out = (float*)d_out;

  // ---- carve workspace (256B aligned slices) ----
  char* base = (char*)d_ws;
  size_t off = 0;
  auto carve = [&](size_t bytes) -> char* {
    off = (off + 255) & ~(size_t)255;
    char* p = base + off;
    off += bytes;
    return p;
  };
  _Float16* eF      = (_Float16*)carve((size_t)EE * KP_G * 2);
  float*    Cc      = (float*)   carve((size_t)EE * 4);
  _Float16* tmp1F   = (_Float16*)carve((size_t)EE * HP * 2);
  _Float16* WtF     = (_Float16*)carve((size_t)EE * HP * 2);
  _Float16* hF      = (_Float16*)carve((size_t)NN * HP * 2);
  float*    h32     = (float*)   carve((size_t)NN * HID * 4);
  float*    x32     = (float*)   carve((size_t)NN * HID * 4);
  _Float16* aggF    = (_Float16*)carve((size_t)NN * HP * 2);
  _Float16* tmp2F   = (_Float16*)carve((size_t)NN * HP * 2);
  _Float16* pooledF = (_Float16*)carve((size_t)GG * HP * 2);
  _Float16* w1F     = (_Float16*)carve((size_t)NIter * HP * KP_G * 2);
  _Float16* w2F     = (_Float16*)carve((size_t)NIter * HP * KP_H * 2);
  _Float16* l1F     = (_Float16*)carve((size_t)NIter * HP * KP_H * 2);
  _Float16* l2F     = (_Float16*)carve((size_t)NIter * HP * KP_H * 2);
  _Float16* lF      = (_Float16*)carve((size_t)NIter * HP * KP_H * 2);
  _Float16* poolF   = (_Float16*)carve((size_t)HP * KP_H * 2);
  if (off > ws_size) return;   // workspace too small: bail safely

  const int cvt_big = (HP * KP_H + 255) / 256;   // 640*640 elements
  const int cvt_sm  = (HP * KP_G + 255) / 256;

  // ---- weight conversion (f32 [K,N] -> f16 [N][K] padded) ----
  for (int t = 0; t < NIter; ++t) {
    convert_wT_kernel<<<cvt_sm, 256, 0, stream>>>(
        mlp_w1 + (size_t)t * NGAU * HID, w1F + (size_t)t * HP * KP_G, NGAU, HID, KP_G, HP);
    convert_wT_kernel<<<cvt_big, 256, 0, stream>>>(
        mlp_w2 + (size_t)t * HID * HID, w2F + (size_t)t * HP * KP_H, HID, HID, KP_H, HP);
    convert_wT_kernel<<<cvt_big, 256, 0, stream>>>(
        lin1_w + (size_t)t * HID * HID, l1F + (size_t)t * HP * KP_H, HID, HID, KP_H, HP);
    convert_wT_kernel<<<cvt_big, 256, 0, stream>>>(
        lin2_w + (size_t)t * HID * HID, l2F + (size_t)t * HP * KP_H, HID, HID, KP_H, HP);
    convert_wT_kernel<<<cvt_big, 256, 0, stream>>>(
        lin_w + (size_t)t * HID * HID, lF + (size_t)t * HP * KP_H, HID, HID, KP_H, HP);
  }
  convert_wT_kernel<<<cvt_big, 256, 0, stream>>>(pool_w, poolF, HID, HID, KP_H, HP);

  // ---- edge features + embedding ----
  edge_feat_kernel<<<(EE + 255) / 256, 256, 0, stream>>>(pos, ei, eF, Cc);
  embed_kernel<<<(NN * HP + 255) / 256, 256, 0, stream>>>(z, emb, h32, hF);

  const int egroups = (EE / 16) * (HP / 64);     // edge-level GEMM wave-groups
  const int ngroups = (NN / 16) * (HP / 64);
  const int pgroups = (GG / 16) * (HP / 64);
  const int eblk = (egroups + 7) / 8;
  const int nblk = (ngroups + 7) / 8;
  const int pblk = (pgroups + 7) / 8;
  const int nelem = (NN * HP + 255) / 256;

  for (int t = 0; t < NIter; ++t) {
    // filter net: ssp(e @ w1 + b1) -> tmp1 (f16)
    gemm_wmma_kernel<<<eblk, 256, 0, stream>>>(
        eF, KP_G, w1F + (size_t)t * HP * KP_G, KP_G,
        mlp_b1 + (size_t)t * HID, nullptr, nullptr,
        nullptr, tmp1F, EE, HID, HP, KP_G, 1);
    // (tmp1 @ w2 + b2) * C -> Wt (f16)    [dominant GEMM: ~51 GFLOP]
    gemm_wmma_kernel<<<eblk, 256, 0, stream>>>(
        tmp1F, HP, w2F + (size_t)t * HP * KP_H, KP_H,
        mlp_b2 + (size_t)t * HID, Cc, nullptr,
        nullptr, WtF, EE, HID, HP, KP_H, 0);
    // x = h @ lin1 (no bias) -> f32
    gemm_wmma_kernel<<<nblk, 256, 0, stream>>>(
        hF, HP, l1F + (size_t)t * HP * KP_H, KP_H,
        nullptr, nullptr, nullptr,
        x32, nullptr, NN, HID, HP, KP_H, 0);
    // agg = scatter_add(x[src] * Wt) via structured reduction
    aggregate_kernel<<<nelem, 256, 0, stream>>>(x32, WtF, aggF);
    // ssp(agg @ lin2 + b) -> tmp2 (f16)
    gemm_wmma_kernel<<<nblk, 256, 0, stream>>>(
        aggF, HP, l2F + (size_t)t * HP * KP_H, KP_H,
        lin2_b + (size_t)t * HID, nullptr, nullptr,
        nullptr, tmp2F, NN, HID, HP, KP_H, 1);
    // h = h + (tmp2 @ lin + b)  -> h32 (f32, in-place residual) and hF (f16)
    gemm_wmma_kernel<<<nblk, 256, 0, stream>>>(
        tmp2F, HP, lF + (size_t)t * HP * KP_H, KP_H,
        lin_b + (size_t)t * HID, nullptr, h32,
        h32, hF, NN, HID, HP, KP_H, 0);
  }

  // ---- mean pool + final linear ----
  pool_kernel<<<(GG * HP + 255) / 256, 256, 0, stream>>>(h32, pooledF);
  gemm_wmma_kernel<<<pblk, 256, 0, stream>>>(
      pooledF, HP, poolF, KP_H,
      pool_b, nullptr, nullptr,
      out, nullptr, GG, HID, HP, KP_H, 0);
}
